// DoubleSin_36146444763907
// MI455X (gfx1250) — matrix operand order repaired and checked by reading, not compiled, not run on hardware
//
#include <hip/hip_runtime.h>

typedef _Float16 v16h __attribute__((ext_vector_type(16)));
typedef _Float16 v8h  __attribute__((ext_vector_type(8)));
typedef _Float16 v2h  __attribute__((ext_vector_type(2)));
typedef float    v8f  __attribute__((ext_vector_type(8)));

#define LDH 72          // padded row stride in halves (144 B: 16B-aligned, bank-conflict-free)
#define TILE_ROWS 256

struct KParams {
    const float* x;
    const float* w[22];   // per branch (11): W1,b1,a1,W2,b2,a2,W3,b3,a3,W4,b4
    float* out;
    int n;
};

// snake(t) = t + sin(a*t)^2/a = (t + h) - h*cos(2*a*t),  h = 1/(2a)
// v_cos_f32 takes revolutions: cos(2*a*t) = cos_rev(t * a/pi), ca = a/pi.
__device__ __forceinline__ float snake_eval(float t, float ca, float h) {
    float c = __builtin_amdgcn_cosf(ca * t);
    return fmaf(-c, h, t + h);
}

// One 64x64 GEMM layer + snake activation for this wave's 32 rows, in-place in sH.
__device__ __forceinline__ void gemm_snake_layer(
    _Float16* __restrict__ sH, const _Float16* __restrict__ sWmat,
    const float* __restrict__ bias, const float* __restrict__ cavec,
    const float* __restrict__ hvec, int rowW, int lo, int hi)
{
    // ---- A fragments: 2 M-tiles x 2 K-fragments (full rows in registers) ----
    v16h A[2][2];
#pragma unroll
    for (int mt = 0; mt < 2; ++mt)
#pragma unroll
        for (int kf = 0; kf < 2; ++kf) {
            const _Float16* pA = &sH[(rowW + mt*16 + lo)*LDH + kf*32 + hi*8];
            v8h a0 = *(const v8h*)pA;
            v8h a1 = *(const v8h*)(pA + 16);
            A[mt][kf] = __builtin_shufflevector(a0, a1,
                0,1,2,3,4,5,6,7,8,9,10,11,12,13,14,15);
        }

    // ---- accumulators initialized with bias (C layout: N = lo for all 8 vgprs) ----
    v8f acc[2][4];
#pragma unroll
    for (int nt = 0; nt < 4; ++nt) {
        float bv = bias[nt*16 + lo];
#pragma unroll
        for (int mt = 0; mt < 2; ++mt)
#pragma unroll
            for (int i = 0; i < 8; ++i) acc[mt][nt][i] = bv;
    }

    // ---- WMMA main loop: 4 n-tiles x 2 k-fragments x 2 m-tiles = 16 WMMAs ----
#pragma unroll
    for (int nt = 0; nt < 4; ++nt)
#pragma unroll
        for (int kf = 0; kf < 2; ++kf) {
            const _Float16* pB = &sWmat[(nt*16 + lo)*LDH + kf*32 + hi*8];
            v8h b0 = *(const v8h*)pB;
            v8h b1 = *(const v8h*)(pB + 16);
            v16h B = __builtin_shufflevector(b0, b1,
                0,1,2,3,4,5,6,7,8,9,10,11,12,13,14,15);
#pragma unroll
            for (int mt = 0; mt < 2; ++mt)
                acc[mt][nt] = __builtin_amdgcn_wmma_f32_16x16x32_f16(
                    false, A[mt][kf], false, B, (short)0, acc[mt][nt],
                    false, false);
        }

    // ---- snake + write back f16 in place ----
#pragma unroll
    for (int nt = 0; nt < 4; ++nt) {
        float ca = cavec[nt*16 + lo];
        float h  = hvec[nt*16 + lo];
#pragma unroll
        for (int mt = 0; mt < 2; ++mt)
#pragma unroll
            for (int i = 0; i < 8; ++i) {
                sH[(rowW + mt*16 + hi*8 + i)*LDH + nt*16 + lo] =
                    (_Float16)snake_eval(acc[mt][nt][i], ca, h);
            }
    }
}

__launch_bounds__(256)
__global__ void snake_mlp_kernel(KParams p)
{
    __shared__ __align__(16) _Float16 sH[TILE_ROWS * LDH];   // activations, 36.9 KB
    __shared__ __align__(16) _Float16 sW[4][64 * LDH];       // W2m1,W3m1,W2m2,W3m2 f16
    __shared__ float sPar[2][11][64];  // W1,b1,ca1,b2,ca2,b3,ca3,W4, h1,h2,h3

    const int tid  = threadIdx.x;
    const int wave = tid >> 5;
    const int lane = tid & 31;
    const int lo   = lane & 15;
    const int hi   = lane >> 4;
    const int rowW = wave * 32;

    // ---- stage weights: f32 global -> f16 LDS, padded [n][LDH] ----
    {
        const float* Wsrc[4] = { p.w[3], p.w[6], p.w[14], p.w[17] };
#pragma unroll
        for (int m = 0; m < 4; ++m)
            for (int idx = tid; idx < 4096; idx += 256) {
                int nn = idx >> 6, kk = idx & 63;
                sW[m][nn*LDH + kk] = (_Float16)Wsrc[m][idx];
            }
        const float* Psrc[2][8] = {
            { p.w[0],  p.w[1],  p.w[2],  p.w[4],  p.w[5],  p.w[7],  p.w[8],  p.w[9]  },
            { p.w[11], p.w[12], p.w[13], p.w[15], p.w[16], p.w[18], p.w[19], p.w[20] } };
        const float INV_PI = 0.31830988618379067f;
        for (int idx = tid; idx < 1408; idx += 256) {
            int br = idx >= 704;
            int rr = idx - br*704;
            int a  = rr >> 6, j = rr & 63;
            float v;
            if (a >= 8)                      v = 0.5f / Psrc[br][(a-8)*2 + 2][j]; // h = 1/(2a)
            else if (a == 2 || a == 4 || a == 6) v = Psrc[br][a][j] * INV_PI;     // ca = a/pi
            else                             v = Psrc[br][a][j];
            sPar[br][a][j] = v;
        }
    }
    const float b4v[2] = { p.w[10][0], p.w[21][0] };
    __syncthreads();

    for (int base = blockIdx.x * TILE_ROWS; base < p.n;
         base += gridDim.x * TILE_ROWS) {
        const int row = base + tid;
        const float xv = (row < p.n) ? p.x[row] : 0.0f;
        float outAcc = 0.0f;

#pragma unroll 1
        for (int br = 0; br < 2; ++br) {
            const float xb = br ? 2.0f * xv : xv;
            const float* W1  = sPar[br][0];
            const float* B1  = sPar[br][1];
            const float* CA1 = sPar[br][2];
            const float* H1  = sPar[br][8];

            // ---- layer 1: h = snake(x*W1 + b1), write row tid as f16 ----
#pragma unroll
            for (int j = 0; j < 64; j += 2) {
                float t0 = fmaf(xb, W1[j],   B1[j]);
                float t1 = fmaf(xb, W1[j+1], B1[j+1]);
                v2h pk = { (_Float16)snake_eval(t0, CA1[j],   H1[j]),
                           (_Float16)snake_eval(t1, CA1[j+1], H1[j+1]) };
                *(v2h*)&sH[tid*LDH + j] = pk;
            }

            // ---- layers 2 & 3: WMMA GEMMs + snake, wave-local, in place ----
            gemm_snake_layer(sH, sW[br*2 + 0], sPar[br][3], sPar[br][4],
                             sPar[br][9],  rowW, lo, hi);
            gemm_snake_layer(sH, sW[br*2 + 1], sPar[br][5], sPar[br][6],
                             sPar[br][10], rowW, lo, hi);

            // ---- layer 4: dot(h, W4) + b4 ----
            const float* W4 = sPar[br][7];
            float dot = b4v[br];
#pragma unroll
            for (int c = 0; c < 8; ++c) {
                v8h hv = *(const v8h*)&sH[tid*LDH + c*8];
#pragma unroll
                for (int i = 0; i < 8; ++i)
                    dot += (float)hv[i] * W4[c*8 + i];
            }
            outAcc += dot;
        }

        if (row < p.n) p.out[row] = outAcc;
    }
}

extern "C" void kernel_launch(void* const* d_in, const int* in_sizes, int n_in,
                              void* d_out, int out_size, void* d_ws, size_t ws_size,
                              hipStream_t stream)
{
    (void)in_sizes; (void)n_in; (void)d_ws; (void)ws_size;
    KParams kp;
    kp.x = (const float*)d_in[0];
    for (int i = 0; i < 22; ++i) kp.w[i] = (const float*)d_in[1 + i];
    kp.out = (float*)d_out;
    kp.n = out_size;

    int tiles = (kp.n + TILE_ROWS - 1) / TILE_ROWS;
    int grid  = tiles < 4096 ? tiles : 4096;
    snake_mlp_kernel<<<dim3(grid), dim3(256), 0, stream>>>(kp);
}
